// EnhancedSNN_35450660061988
// MI455X (gfx1250) — matrix operand order repaired and checked by reading, not compiled, not run on hardware
//
#include <hip/hip_runtime.h>

// ---------------------------------------------------------------------------
// EnhancedSNN for MI455X (gfx1250).
// Dominant cost: 32 sequential steps of s1[2048x4096] @ W2^T[4096x1024]
// (~550 GFLOP of f16 WMMA). Working set (~116 MB) is L2-resident; only the
// spike tensor (268 MB) streams to HBM, stored nontemporally.
// GEMM: WG = 128 threads (4 waves), WG tile 128x64, wave tile 32x64
// (2x4 v_wmma_f32_16x16x32_f16 accumulators). W2 tile is staged into LDS
// (double-buffered, GLOBAL_LOAD_ASYNC_TO_LDS_B128) and shared by all 4 waves.
// ---------------------------------------------------------------------------

typedef __attribute__((ext_vector_type(16))) _Float16 v16h;
typedef __attribute__((ext_vector_type(8)))  _Float16 h8;
typedef __attribute__((ext_vector_type(4)))  _Float16 h4;
typedef __attribute__((ext_vector_type(8)))  float    v8f;
typedef __attribute__((ext_vector_type(4)))  float    f4;
typedef __attribute__((ext_vector_type(4)))  int      i4;

#define B_     2048
#define IN_    1024
#define HID_   4096
#define OUT_   1024
#define TSTEPS 32
#define THR1   0.8f
#define THR2   0.5f
#define LEAK   0.9f

#define KC     32          // K elements per staged chunk
#define BROWS  64          // staged B rows per chunk (= WG tile N)
#define BPAD   40          // f16 stride per LDS row: 32 data + 8 pad (80 B)

#if __has_builtin(__builtin_amdgcn_global_load_async_to_lds_b128)
#define HAVE_ASYNC_LDS 1
// Builtin signature (from clang diagnostic): arg0 is int4 in AS(1) ("__device__"),
// arg1 expected to be the LDS destination in AS(3) ("__shared__").
typedef __attribute__((address_space(1))) i4* as1_i4p;
typedef __attribute__((address_space(3))) i4* as3_i4p;
#else
#define HAVE_ASYNC_LDS 0
#endif

__device__ __forceinline__ void async_wait0() {
#if __has_builtin(__builtin_amdgcn_s_wait_asynccnt)
    __builtin_amdgcn_s_wait_asynccnt(0);
#else
    asm volatile("s_wait_asynccnt 0x0" ::: "memory");
#endif
}

// ------------------------------ helpers ------------------------------------

__global__ void snn_cvt_f16(const float* __restrict__ in, _Float16* __restrict__ out, int n) {
    int i = (blockIdx.x * 256 + threadIdx.x) * 4;
    if (i >= n) return;
    f4 v = *(const f4*)(in + i);
    h4 r;
    r[0] = (_Float16)v[0]; r[1] = (_Float16)v[1];
    r[2] = (_Float16)v[2]; r[3] = (_Float16)v[3];
    *(h4*)(out + i) = r;
}

__global__ void snn_zero(float* __restrict__ p, int n) {
    int i = (blockIdx.x * 256 + threadIdx.x) * 4;
    if (i >= n) return;
    *(f4*)(p + i) = (f4){0.0f, 0.0f, 0.0f, 0.0f};
}

// LIF layer 1: v1 = v1*LEAK + current1; spike if v1>=THR1; reset on spike.
// Spikes written as f16 (exact for {0,1}) feeding the WMMA GEMM.
__global__ void snn_lif1(float* __restrict__ v1, const float* __restrict__ cur1,
                         _Float16* __restrict__ s1, int n) {
    int i = (blockIdx.x * 256 + threadIdx.x) * 4;
    if (i >= n) return;
    f4 v = *(const f4*)(v1 + i);
    f4 c = *(const f4*)(cur1 + i);
    h4 s;
#pragma unroll
    for (int j = 0; j < 4; ++j) {
        float vv = v[j] * LEAK + c[j];
        float sp = (vv >= THR1) ? 1.0f : 0.0f;
        v[j] = (sp != 0.0f) ? 0.0f : vv;
        s[j] = (_Float16)sp;
    }
    *(f4*)(v1 + i) = v;
    *(h4*)(s1 + i) = s;
}

// A-fragment (16x32 f16, MxK): lanes 0-15 row M=lane, K = k0+{0..7,16..23};
// lanes 16-31 same rows, K = k0+{8..15,24..31}. Two b128 loads per lane.
__device__ __forceinline__ v16h ld_fragA(const _Float16* __restrict__ p) {
    h8 lo = *(const h8*)(p);
    h8 hi = *(const h8*)(p + 16);
    v16h r;
#pragma unroll
    for (int i = 0; i < 8; ++i) { r[i] = lo[i]; r[i + 8] = hi[i]; }
    return r;
}

// ------------------------------ WMMA GEMM ----------------------------------
// C[M,N] = A[M,K] @ Bt[N,K]^T + bias[N].
// EPI==0: store C (current1).  EPI==1: fused LIF2 + spike emit + rate accum.
template <int EPI>
__global__ __launch_bounds__(128) void snn_gemm_wmma(
    const _Float16* __restrict__ A, const _Float16* __restrict__ Bt,
    const float* __restrict__ bias,
    float* __restrict__ Cout,                            // EPI==0
    float* __restrict__ v2, float* __restrict__ spikes,  // EPI==1 (spikes pre-offset by t*OUT)
    float* __restrict__ ssum,                            // EPI==1
    int M, int N, int K)
{
    __shared__ _Float16 ldsB[2][BROWS][BPAD];

    const int wave   = threadIdx.x >> 5;          // 0..3 (M direction)
    const int lane   = threadIdx.x & 31;
    const int m0     = blockIdx.x * 128 + wave * 32;
    const int n0     = blockIdx.y * 64;
    const int laneLo = lane & 15;
    const int halfHi = lane >> 4;                 // 0 or 1

    // Direct-global A fragment pointers (two 16-row fragments per wave).
    const _Float16* a0 = A + (size_t)(m0 + laneLo) * K + halfHi * 8;
    const _Float16* a1 = a0 + (size_t)16 * K;

    // B staging: wave w stages rows [w*16, w*16+16) of the 64-row tile.
    // One async b128 covers 8 rows (4 lanes x 16B per row).
    const int srow = wave * 16 + (lane >> 2);     // + 0 and + 8
    const int scol = (lane & 3) * 8;              // f16 offset within 32-K chunk
    const _Float16* bsrc = Bt + (size_t)(n0 + srow) * K + scol;

    v8f acc[2][4] = {};

    auto issueB = [&](int bsel, int k0) {
#if HAVE_ASYNC_LDS
        __builtin_amdgcn_global_load_async_to_lds_b128(
            (as1_i4p)(void*)(bsrc + k0),
            (as3_i4p)(void*)&ldsB[bsel][srow][scol], 0, 0);
        __builtin_amdgcn_global_load_async_to_lds_b128(
            (as1_i4p)(void*)(bsrc + (size_t)8 * K + k0),
            (as3_i4p)(void*)&ldsB[bsel][srow + 8][scol], 0, 0);
#else
        *(h8*)&ldsB[bsel][srow][scol]     = *(const h8*)(bsrc + k0);
        *(h8*)&ldsB[bsel][srow + 8][scol] = *(const h8*)(bsrc + (size_t)8 * K + k0);
#endif
    };

    // B-fragment from LDS (32x16 f16, KxN): lanes 0-15 col N=f*16+lane, K {0..7,16..23};
    // lanes 16-31 same cols, K {8..15,24..31}. Two ds b128 loads per lane, 80 B row
    // stride keeps the wave at the 2-pass bank minimum.
    auto ldB = [&](int bsel, int f) -> v16h {
        const _Float16* p = &ldsB[bsel][f * 16 + laneLo][halfHi * 8];
        h8 lo = *(const h8*)(p);
        h8 hi = *(const h8*)(p + 16);
        v16h r;
#pragma unroll
        for (int i = 0; i < 8; ++i) { r[i] = lo[i]; r[i + 8] = hi[i]; }
        return r;
    };

    // Prologue: stage chunk 0.
    issueB(0, 0);
#if HAVE_ASYNC_LDS
    async_wait0();
#endif
    __syncthreads();

    const int NKC = K / KC;
    for (int kc = 0; kc < NKC; ++kc) {
        const int k0  = kc * KC;
        const int cur = kc & 1;
        if (kc + 1 < NKC) issueB(cur ^ 1, k0 + KC);   // prefetch next chunk into LDS

        __builtin_prefetch(a0 + k0 + 128, 0, 1);      // global_prefetch_b8 for A stream
        v16h fa0 = ld_fragA(a0 + k0);
        v16h fa1 = ld_fragA(a1 + k0);
#pragma unroll
        for (int f = 0; f < 4; ++f) {
            v16h fb = ldB(cur, f);
            acc[0][f] = __builtin_amdgcn_wmma_f32_16x16x32_f16(false, fa0, false, fb,
                                                               (short)0, acc[0][f], false, false);
            acc[1][f] = __builtin_amdgcn_wmma_f32_16x16x32_f16(false, fa1, false, fb,
                                                               (short)0, acc[1][f], false, false);
        }
        if (kc + 1 < NKC) {
#if HAVE_ASYNC_LDS
            async_wait0();                             // my async writes for next chunk done
#endif
            __syncthreads();                           // all waves: next buffer ready, cur reads drained
        }
    }

    // Epilogue. C/D f32 layout: VGPR i, lanes 0-15 -> row M=i, lanes 16-31 -> M=i+8.
#pragma unroll
    for (int mi = 0; mi < 2; ++mi) {
#pragma unroll
        for (int f = 0; f < 4; ++f) {
            const int col  = n0 + f * 16 + laneLo;
            const int row0 = m0 + mi * 16 + halfHi * 8;
            const float bv = bias[col];
            v8f a = acc[mi][f];
#pragma unroll
            for (int i = 0; i < 8; ++i) {
                const int row = row0 + i;
                const float c = a[i] + bv;
                if (EPI == 0) {
                    Cout[(size_t)row * N + col] = c;
                } else {
                    const size_t idx = (size_t)row * N + col;
                    float v = v2[idx] * LEAK + c;
                    float s = (v >= THR2) ? 1.0f : 0.0f;
                    v2[idx] = (s != 0.0f) ? 0.0f : v;
                    // Spike tensor is the only HBM stream (268 MB): keep it out of L2.
                    __builtin_nontemporal_store(s, &spikes[(size_t)row * (TSTEPS * OUT_) + col]);
                    ssum[idx] += s;
                }
            }
        }
    }
}

// Readout: out[b, j] = (ssum[b,:]/T) . Wr[j,:] + br[j], j in {0,1}.
__global__ void snn_readout(const float* __restrict__ ssum, const float* __restrict__ Wr,
                            const float* __restrict__ br, float* __restrict__ out) {
    __shared__ float red0[256];
    __shared__ float red1[256];
    const int b = blockIdx.x;
    const int tid = threadIdx.x;
    float s0 = 0.0f, s1 = 0.0f;
    for (int o = tid; o < OUT_; o += 256) {
        const float r = ssum[(size_t)b * OUT_ + o] * (1.0f / (float)TSTEPS);
        s0 += r * Wr[o];
        s1 += r * Wr[OUT_ + o];
    }
    red0[tid] = s0; red1[tid] = s1;
    __syncthreads();
    for (int st = 128; st > 0; st >>= 1) {
        if (tid < st) { red0[tid] += red0[tid + st]; red1[tid] += red1[tid + st]; }
        __syncthreads();
    }
    if (tid == 0) {
        out[b * 2 + 0] = red0[0] + br[0];
        out[b * 2 + 1] = red1[0] + br[1];
    }
}

// ------------------------------ launcher -----------------------------------

extern "C" void kernel_launch(void* const* d_in, const int* in_sizes, int n_in,
                              void* d_out, int out_size, void* d_ws, size_t ws_size,
                              hipStream_t stream) {
    const float* x  = (const float*)d_in[0];   // [B, IN]
    const float* W1 = (const float*)d_in[1];   // [HID, IN]
    const float* b1 = (const float*)d_in[2];   // [HID]
    const float* W2 = (const float*)d_in[3];   // [OUT, HID]
    const float* b2 = (const float*)d_in[4];   // [OUT]
    const float* Wr = (const float*)d_in[5];   // [2, OUT]
    const float* br = (const float*)d_in[6];   // [2]
    // d_in[7] = num_steps (fixed at 32 per reference setup)

    float* out          = (float*)d_out;           // [B, 2]
    float* spike_tensor = out + (size_t)B_ * 2;    // [B, T, OUT]

    // Workspace layout (~116 MB, fits MI455X 192 MB L2 as hot set).
    char* ws = (char*)d_ws;
    size_t off = 0;
    auto carve = [&](size_t bytes) { char* p = ws + off; off += (bytes + 255) & ~(size_t)255; return p; };
    _Float16* xh   = (_Float16*)carve((size_t)B_  * IN_  * 2);
    _Float16* W1h  = (_Float16*)carve((size_t)HID_ * IN_  * 2);
    _Float16* W2h  = (_Float16*)carve((size_t)OUT_ * HID_ * 2);
    _Float16* s1h  = (_Float16*)carve((size_t)B_  * HID_ * 2);
    float*    cur1 = (float*)   carve((size_t)B_  * HID_ * 4);
    float*    v1   = (float*)   carve((size_t)B_  * HID_ * 4);
    float*    v2   = (float*)   carve((size_t)B_  * OUT_ * 4);
    float*    ssum = (float*)   carve((size_t)B_  * OUT_ * 4);

    // f32 -> f16 conversions (once per call).
    snn_cvt_f16<<<(B_  * IN_ ) / 1024, 256, 0, stream>>>(x,  xh,  B_  * IN_);
    snn_cvt_f16<<<(HID_ * IN_) / 1024, 256, 0, stream>>>(W1, W1h, HID_ * IN_);
    snn_cvt_f16<<<(OUT_ * HID_) / 1024, 256, 0, stream>>>(W2, W2h, OUT_ * HID_);

    // Zero state (workspace is poisoned; must be deterministic each call).
    snn_zero<<<(B_ * HID_) / 1024, 256, 0, stream>>>(v1,   B_ * HID_);
    snn_zero<<<(B_ * OUT_) / 1024, 256, 0, stream>>>(v2,   B_ * OUT_);
    snn_zero<<<(B_ * OUT_) / 1024, 256, 0, stream>>>(ssum, B_ * OUT_);

    // current1 = x @ W1^T + b1   (M=2048, N=4096, K=1024)
    {
        dim3 grid(B_ / 128, HID_ / 64);
        snn_gemm_wmma<0><<<grid, 128, 0, stream>>>(xh, W1h, b1, cur1,
                                                   nullptr, nullptr, nullptr,
                                                   B_, HID_, IN_);
    }

    // 32 recurrent steps: LIF1 -> WMMA GEMM (s1 @ W2^T) fused with LIF2.
    for (int t = 0; t < TSTEPS; ++t) {
        snn_lif1<<<(B_ * HID_) / 1024, 256, 0, stream>>>(v1, cur1, s1h, B_ * HID_);
        dim3 grid(B_ / 128, OUT_ / 64);
        snn_gemm_wmma<1><<<grid, 128, 0, stream>>>(s1h, W2h, b2, nullptr,
                                                   v2, spike_tensor + (size_t)t * OUT_, ssum,
                                                   B_, OUT_, HID_);
    }

    // output = spike_rate @ Wr^T + br
    snn_readout<<<B_, 256, 0, stream>>>(ssum, Wr, br, out);
}
